// RelPosSelfAttention_53695681135021
// MI455X (gfx1250) — matrix operand order repaired and checked by reading, not compiled, not run on hardware
//
#include <hip/hip_runtime.h>
#include <cstddef>
#include <cstdint>

#define D_MODEL 512
#define NHEAD 8
#define HEAD_DIM 64
#define NS 1
#define BATCH 8
#define SEQ 1024
#define TOKENS (BATCH * SEQ)

typedef __attribute__((ext_vector_type(16))) __bf16 bf16x16;
typedef __attribute__((ext_vector_type(8))) float floatx8;

union Frag16 {
    bf16x16 v;
    uint4 q[2];
    unsigned short s[16];
};

__device__ __forceinline__ unsigned short f2bf(float f) {
    unsigned int u = __float_as_uint(f);
    u += 0x7fffu + ((u >> 16) & 1u);
    return (unsigned short)(u >> 16);
}
__device__ __forceinline__ float bf2f(unsigned short u) {
    return __uint_as_float(((unsigned int)u) << 16);
}
__device__ __forceinline__ floatx8 wmma_bf16(bf16x16 a, bf16x16 b, floatx8 c) {
    // D = A(16x32) * B(32x16) + C, fp32 accumulate
    return __builtin_amdgcn_wmma_f32_16x16x32_bf16(false, a, false, b, (short)0, c, false, false);
}

// Async global->LDS copy (CDNA5): per-lane 16B, tracked by ASYNCcnt.
// For generic pointers to LDS, the low 32 bits are the LDS byte offset.
__device__ __forceinline__ void async_ld_b128(void* lds_ptr, const void* gptr) {
    unsigned loff = (unsigned)(uintptr_t)lds_ptr;
    unsigned long long ga = (unsigned long long)(uintptr_t)gptr;
    asm volatile("global_load_async_to_lds_b128 %0, %1, off" ::"v"(loff), "v"(ga)
                 : "memory");
}
__device__ __forceinline__ void wait_async0() {
    asm volatile("s_wait_asynccnt 0" ::: "memory");
}

// ---------------------------------------------------------------------------
// fp32 -> bf16 conversion (vectorized by 4)
// ---------------------------------------------------------------------------
__global__ __launch_bounds__(256) void cvt_f32_bf16(const float* __restrict__ in,
                                                    unsigned short* __restrict__ out, int n) {
    int i = (blockIdx.x * 256 + threadIdx.x) * 4;
    if (i + 3 < n) {
        float4 v = *(const float4*)(in + i);
        unsigned long long p = (unsigned long long)f2bf(v.x) |
                               ((unsigned long long)f2bf(v.y) << 16) |
                               ((unsigned long long)f2bf(v.z) << 32) |
                               ((unsigned long long)f2bf(v.w) << 48);
        *(unsigned long long*)(out + i) = p;
    } else {
        for (; i < n; ++i) out[i] = f2bf(in[i]);
    }
}

// ---------------------------------------------------------------------------
// WMMA bf16 GEMM:  C[M,Ncols] = A[M,K] * W[Ncols,K]^T   (both K-contiguous)
// 256 threads (8 waves). Block tile 128x64; wave tile 32x32 (4 WMMA accs).
// Two-stage async global->LDS pipeline (ASYNCcnt), K chunked by 32.
// EPI==0: scatter into Q(scaled)/K/V bf16 [B,H,N,dh].  EPI==1: fp32 + bias.
// ---------------------------------------------------------------------------
template <int EPI>
__global__ __launch_bounds__(256) void gemm_bf16_k(
    const unsigned short* __restrict__ A, const unsigned short* __restrict__ W,
    const float* __restrict__ obias, float* __restrict__ outF,
    unsigned short* __restrict__ Qo, unsigned short* __restrict__ Ko,
    unsigned short* __restrict__ Vo, int K, int Ncols, float qscale) {
    __shared__ unsigned short sA[2][128 * 40];
    __shared__ unsigned short sW[2][64 * 40];
    const int tid = threadIdx.x;
    const int lane = tid & 31, wave = tid >> 5;
    const int wm = wave & 3, wn = wave >> 2;     // wave tile: rows wm*32, cols wn*32
    const int M0 = blockIdx.y * 128, N0 = blockIdx.x * 64;

    const floatx8 zero8 = {0.f, 0.f, 0.f, 0.f, 0.f, 0.f, 0.f, 0.f};
    floatx8 acc[2][2] = {{zero8, zero8}, {zero8, zero8}};

    // cooperative loader geometry
    const int arow = tid >> 1, aseg = (tid & 1) << 4;  // 128 rows x 32B chunks
    const int wrow = tid >> 2, wseg = (tid & 3) << 3;  // 64 rows x 16B chunks
    const unsigned short* aBase = A + (size_t)(M0 + arow) * K + aseg;
    const unsigned short* wBase = W + (size_t)(N0 + wrow) * K + wseg;

    auto issue = [&](int buf, int kk) {
        async_ld_b128(&sA[buf][arow * 40 + aseg], aBase + kk);
        async_ld_b128(&sA[buf][arow * 40 + aseg + 8], aBase + kk + 8);
        async_ld_b128(&sW[buf][wrow * 40 + wseg], wBase + kk);
    };
    auto stage = [&](int buf) {
        Frag16 a[2], b[2];
        const int ab = (lane >> 4) << 3;
        const int kb = (lane >> 4) << 3;
#pragma unroll
        for (int mi = 0; mi < 2; ++mi) {
            const int m = wm * 32 + mi * 16 + (lane & 15);
            a[mi].q[0] = *(const uint4*)(&sA[buf][m * 40 + ab]);
            a[mi].q[1] = *(const uint4*)(&sA[buf][m * 40 + ab + 16]);
        }
#pragma unroll
        for (int ni = 0; ni < 2; ++ni) {
            const int n = wn * 32 + ni * 16 + (lane & 15);
            b[ni].q[0] = *(const uint4*)(&sW[buf][n * 40 + kb]);
            b[ni].q[1] = *(const uint4*)(&sW[buf][n * 40 + kb + 16]);
        }
#pragma unroll
        for (int mi = 0; mi < 2; ++mi)
#pragma unroll
            for (int ni = 0; ni < 2; ++ni)
                acc[mi][ni] = wmma_bf16(a[mi].v, b[ni].v, acc[mi][ni]);
    };

    // prologue: async-fill stage 0
    issue(0, 0);
    wait_async0();
    __syncthreads();

    for (int k0 = 0; k0 < K; k0 += 64) {  // K multiple of 64
        // even stage: compute buf0 while async-filling buf1 with k0+32
        if (k0 + 32 < K) issue(1, k0 + 32);
        stage(0);
        wait_async0();
        __syncthreads();
        // odd stage: compute buf1 while async-filling buf0 with k0+64
        if (k0 + 64 < K) issue(0, k0 + 64);
        stage(1);
        wait_async0();
        __syncthreads();
    }

    const int nlo = lane & 15, mhi = (lane >> 4) * 8;
#pragma unroll
    for (int mi = 0; mi < 2; ++mi) {
#pragma unroll
        for (int ni = 0; ni < 2; ++ni) {
            const int col = N0 + wn * 32 + ni * 16 + nlo;
#pragma unroll
            for (int r = 0; r < 8; ++r) {
                const int row = M0 + wm * 32 + mi * 16 + r + mhi;
                const float v = acc[mi][ni][r];
                if (EPI == 0) {
                    const int b = row >> 10, n = row & 1023;
                    const int h = (col >> 6) & 7, d = col & 63;
                    const size_t idx =
                        (((size_t)(b * NHEAD + h)) * SEQ + n) * HEAD_DIM + d;
                    if (col < D_MODEL)          Qo[idx] = f2bf(v * qscale);
                    else if (col < 2 * D_MODEL) Ko[idx] = f2bf(v);
                    else                        Vo[idx] = f2bf(v);
                } else {
                    outF[(size_t)row * Ncols + col] = v + obias[col];
                }
            }
        }
    }
}

// ---------------------------------------------------------------------------
// Fused flash attention with pairwise coordinate-MLP bias.
// Grid: (N/16, B). Block: 256 threads = 8 waves; wave h handles head h.
// Bias MLP computed once per (i,j) pair (shared by all heads via LDS),
// stored [h][j][m] bf16 so each wave reads its slice with one b128 per tile.
// ---------------------------------------------------------------------------
__global__ __launch_bounds__(256) void attn_kernel(
    const unsigned short* __restrict__ Qb, const unsigned short* __restrict__ Kb,
    const unsigned short* __restrict__ Vb, const float* __restrict__ coords,
    const unsigned char* __restrict__ kpm, const float* __restrict__ w1,
    const float* __restrict__ b1, const float* __restrict__ w2,
    const float* __restrict__ b2, const float* __restrict__ alpha_p,
    const float* __restrict__ cscale, unsigned short* __restrict__ attn_out) {
    __shared__ unsigned short s_bias[NHEAD * 32 * 16];  // bf16 [h][j(32)][m(16)]
    __shared__ unsigned short s_vt[NHEAD][64 * 40];     // V tile transposed [d][j]
    __shared__ unsigned short s_p[NHEAD][16 * 40];      // P tile [m][j]
    __shared__ float s_ci[16 * 3];
    __shared__ float s_cj[32 * 3];
    __shared__ float s_w1[32 * 4], s_b1[32], s_w2[8 * 32], s_b2[8];

    const int tid = threadIdx.x, lane = tid & 31, h = tid >> 5;
    const int bz = blockIdx.y;
    const int i0 = blockIdx.x * 16;
    const float alpha = alpha_p[0];

    if (tid < 128) s_w1[tid] = w1[tid];
    if (tid < 32) s_b1[tid] = b1[tid];
    s_w2[tid] = w2[tid];  // exactly 256
    if (tid < 8) s_b2[tid] = b2[tid];
    if (tid < 48)
        s_ci[tid] = coords[((size_t)bz * SEQ + i0 + tid / 3) * 3 + tid % 3] /
                    cscale[tid % 3];

    // Q fragments (softmax scale folded in at QKV-projection time)
    Frag16 qa[2];
    {
        const size_t qbase =
            (((size_t)(bz * NHEAD + h)) * SEQ + i0 + (lane & 15)) * HEAD_DIM;
        const int off = (lane >> 4) << 3;
#pragma unroll
        for (int c = 0; c < 2; ++c) {
            qa[c].q[0] = *(const uint4*)(Qb + qbase + c * 32 + off);
            qa[c].q[1] = *(const uint4*)(Qb + qbase + c * 32 + off + 16);
        }
    }

    const floatx8 zero8 = {0.f, 0.f, 0.f, 0.f, 0.f, 0.f, 0.f, 0.f};
    floatx8 o[4] = {zero8, zero8, zero8, zero8};
    float mrow[8], lsum[8];
#pragma unroll
    for (int r = 0; r < 8; ++r) { mrow[r] = -1e30f; lsum[r] = 0.f; }

    const size_t hseq = ((size_t)(bz * NHEAD + h)) * SEQ;

    for (int j0 = 0; j0 < SEQ; j0 += 32) {
        __syncthreads();  // previous iteration fully consumed s_bias
        if (tid < 96)
            s_cj[tid] = coords[((size_t)bz * SEQ + j0 + tid / 3) * 3 + tid % 3] /
                        cscale[tid % 3];
        __syncthreads();

        // prefetch next j-tile of K and V (global_prefetch_b8)
        if (j0 + 32 < SEQ) {
            __builtin_prefetch(Kb + (hseq + j0 + 32 + lane) * HEAD_DIM, 0, 3);
            __builtin_prefetch(Vb + (hseq + j0 + 32 + lane) * HEAD_DIM, 0, 3);
        }

        // ---- pairwise bias MLP: 512 pairs, 2 per thread, all heads at once ----
#pragma unroll
        for (int pp = 0; pp < 2; ++pp) {
            const int p = tid + pp * 256;
            const int il = p >> 5, jl = p & 31;
            const bool zo = (i0 + il < NS) || (j0 + jl < NS);
            const float dx = s_ci[il * 3 + 0] - s_cj[jl * 3 + 0];
            const float dy = s_ci[il * 3 + 1] - s_cj[jl * 3 + 1];
            const float dz = s_ci[il * 3 + 2] - s_cj[jl * 3 + 2];
            const float dist = sqrtf(dx * dx + dy * dy + dz * dz);
            float acc[8] = {0.f, 0.f, 0.f, 0.f, 0.f, 0.f, 0.f, 0.f};
#pragma unroll 4
            for (int kk = 0; kk < 32; ++kk) {
                float hv = s_b1[kk] + s_w1[kk * 4 + 0] * dx +
                           s_w1[kk * 4 + 1] * dy + s_w1[kk * 4 + 2] * dz +
                           s_w1[kk * 4 + 3] * dist;
                float g = 0.5f * hv * (1.0f + erff(hv * 0.70710678118f));
#pragma unroll
                for (int h2 = 0; h2 < 8; ++h2) acc[h2] += s_w2[h2 * 32 + kk] * g;
            }
            unsigned short* dst = &s_bias[jl * 16 + il];  // [h][j][m], h-stride 512
#pragma unroll
            for (int h2 = 0; h2 < 8; ++h2)
                dst[h2 * 512] = f2bf(zo ? 0.f : alpha * (acc[h2] + s_b2[h2]));
        }

        // ---- stage V tile transposed: lane stages two adjacent j-rows of a
        //      32-wide d-half and packs row pairs into b32 stores ----
        {
            const int g = lane & 15, dbase = (lane >> 4) * 32;
            union { uint4 q[4]; unsigned short s[32]; } r0, r1;
            const size_t vb0 = (hseq + j0 + 2 * g) * HEAD_DIM + dbase;
#pragma unroll
            for (int q4 = 0; q4 < 4; ++q4) {
                r0.q[q4] = *(const uint4*)(Vb + vb0 + q4 * 8);
                r1.q[q4] = *(const uint4*)(Vb + vb0 + HEAD_DIM + q4 * 8);
            }
#pragma unroll
            for (int dk = 0; dk < 32; ++dk) {
                unsigned pk = (unsigned)r0.s[dk] | ((unsigned)r1.s[dk] << 16);
                *(unsigned*)(&s_vt[h][(dbase + dk) * 40 + 2 * g]) = pk;
            }
        }
        __syncthreads();

        // ---- S = Q K^T (+bias, +padding mask) : two 16x16 WMMA tiles ----
        floatx8 sfr[2];
#pragma unroll
        for (int jt = 0; jt < 2; ++jt) {
            floatx8 acc = zero8;
            const int jl = jt * 16 + (lane & 15);
            const size_t kbase = (hseq + j0 + jl) * HEAD_DIM;
#pragma unroll
            for (int c = 0; c < 2; ++c) {
                Frag16 kb;
                const int off = c * 32 + ((lane >> 4) << 3);
                kb.q[0] = *(const uint4*)(Kb + kbase + off);
                kb.q[1] = *(const uint4*)(Kb + kbase + off + 16);
                acc = wmma_bf16(qa[c].v, kb.v, acc);
            }
            // one 16B LDS read delivers this lane's 8 bias rows for column jl
            union { uint4 q; unsigned short s[8]; } bfr;
            bfr.q = *(const uint4*)(&s_bias[h * 512 + jl * 16 + ((lane >> 4) << 3)]);
            const unsigned char mk = kpm[bz * SEQ + j0 + jl];
#pragma unroll
            for (int r = 0; r < 8; ++r) {
                float lg = acc[r] + bf2f(bfr.s[r]);
                if (mk) lg = -1e30f;
                acc[r] = lg;
            }
            sfr[jt] = acc;
        }

        // ---- online softmax (row groups live within 16-lane halves) ----
#pragma unroll
        for (int r = 0; r < 8; ++r) {
            float mloc = fmaxf(sfr[0][r], sfr[1][r]);
            mloc = fmaxf(mloc, __shfl_xor(mloc, 1, 32));
            mloc = fmaxf(mloc, __shfl_xor(mloc, 2, 32));
            mloc = fmaxf(mloc, __shfl_xor(mloc, 4, 32));
            mloc = fmaxf(mloc, __shfl_xor(mloc, 8, 32));
            const float mnew = fmaxf(mrow[r], mloc);
            const float sc = __expf(mrow[r] - mnew);
            const float p0 = __expf(sfr[0][r] - mnew);
            const float p1 = __expf(sfr[1][r] - mnew);
            float rs = p0 + p1;
            rs += __shfl_xor(rs, 1, 32);
            rs += __shfl_xor(rs, 2, 32);
            rs += __shfl_xor(rs, 4, 32);
            rs += __shfl_xor(rs, 8, 32);
            lsum[r] = lsum[r] * sc + rs;
            mrow[r] = mnew;
            o[0][r] *= sc; o[1][r] *= sc; o[2][r] *= sc; o[3][r] *= sc;
            const int m = r + ((lane >> 4) << 3);
            s_p[h][m * 40 + (lane & 15)] = f2bf(p0);
            s_p[h][m * 40 + 16 + (lane & 15)] = f2bf(p1);
        }
        __builtin_amdgcn_wave_barrier();  // DS ops are in-order per wave

        // ---- O += P V : A-frag from per-wave P tile, B-frags from Vt ----
        Frag16 pa;
        {
            const int m = lane & 15;
            const int ab = (lane >> 4) << 3;
            pa.q[0] = *(const uint4*)(&s_p[h][m * 40 + ab]);
            pa.q[1] = *(const uint4*)(&s_p[h][m * 40 + ab + 16]);
        }
#pragma unroll
        for (int f = 0; f < 4; ++f) {
            Frag16 vb;
            const int d = f * 16 + (lane & 15);
            const int kb = (lane >> 4) << 3;
            vb.q[0] = *(const uint4*)(&s_vt[h][d * 40 + kb]);
            vb.q[1] = *(const uint4*)(&s_vt[h][d * 40 + kb + 16]);
            o[f] = wmma_bf16(pa.v, vb.v, o[f]);
        }
    }

    // ---- normalize and store attention output as bf16 [B,N,H*dh] ----
    const size_t obase = ((size_t)bz * SEQ) * D_MODEL;
#pragma unroll
    for (int f = 0; f < 4; ++f) {
#pragma unroll
        for (int r = 0; r < 8; ++r) {
            const int i = i0 + r + ((lane >> 4) << 3);
            const int d = h * 64 + f * 16 + (lane & 15);
            attn_out[obase + (size_t)i * D_MODEL + d] = f2bf(o[f][r] / lsum[r]);
        }
    }
}

// ---------------------------------------------------------------------------
extern "C" void kernel_launch(void* const* d_in, const int* in_sizes, int n_in,
                              void* d_out, int out_size, void* d_ws,
                              size_t ws_size, hipStream_t stream) {
    (void)in_sizes; (void)n_in; (void)out_size; (void)ws_size;
    const float* x = (const float*)d_in[0];
    const float* coords = (const float*)d_in[1];
    const unsigned char* kpm = (const unsigned char*)d_in[2];
    const float* qkv_w = (const float*)d_in[3];
    const float* out_w = (const float*)d_in[4];
    const float* out_b = (const float*)d_in[5];
    const float* bias_alpha = (const float*)d_in[6];
    const float* w1 = (const float*)d_in[7];
    const float* b1 = (const float*)d_in[8];
    const float* w2 = (const float*)d_in[9];
    const float* b2 = (const float*)d_in[10];
    const float* cscale = (const float*)d_in[11];
    float* out = (float*)d_out;

    char* w = (char*)d_ws;
    auto alloc = [&](size_t bytes) -> void* {
        void* p = (void*)w;
        w += (bytes + 255) & ~(size_t)255;
        return p;
    };
    unsigned short* xb = (unsigned short*)alloc((size_t)TOKENS * D_MODEL * 2);
    unsigned short* qkvwb = (unsigned short*)alloc((size_t)3 * D_MODEL * D_MODEL * 2);
    unsigned short* outwb = (unsigned short*)alloc((size_t)D_MODEL * D_MODEL * 2);
    const size_t hsz = (size_t)BATCH * NHEAD * SEQ * HEAD_DIM;
    unsigned short* Qb = (unsigned short*)alloc(hsz * 2);
    unsigned short* Kb = (unsigned short*)alloc(hsz * 2);
    unsigned short* Vb = (unsigned short*)alloc(hsz * 2);
    unsigned short* ao = (unsigned short*)alloc((size_t)TOKENS * D_MODEL * 2);

    // bf16 down-conversion of activations and weights
    cvt_f32_bf16<<<(TOKENS * D_MODEL) / 1024, 256, 0, stream>>>(x, xb,
                                                                TOKENS * D_MODEL);
    cvt_f32_bf16<<<(3 * D_MODEL * D_MODEL) / 1024, 256, 0, stream>>>(
        qkv_w, qkvwb, 3 * D_MODEL * D_MODEL);
    cvt_f32_bf16<<<(D_MODEL * D_MODEL) / 1024, 256, 0, stream>>>(
        out_w, outwb, D_MODEL * D_MODEL);

    // QKV projection: [8192,512] x [1536,512]^T, scatter to Q(scaled)/K/V
    gemm_bf16_k<0><<<dim3(24, 64), 256, 0, stream>>>(
        xb, qkvwb, nullptr, nullptr, Qb, Kb, Vb, D_MODEL, 3 * D_MODEL, 0.125f);

    // Fused flash attention + pairwise bias MLP
    attn_kernel<<<dim3(SEQ / 16, BATCH), 256, 0, stream>>>(
        Qb, Kb, Vb, coords, kpm, w1, b1, w2, b2, bias_alpha, cscale, ao);

    // Output projection: [8192,512] x [512,512]^T + out_b -> fp32
    gemm_bf16_k<1><<<dim3(8, 64), 256, 0, stream>>>(
        ao, outwb, out_b, out, nullptr, nullptr, nullptr, D_MODEL, D_MODEL, 1.f);
}
